// PointsToBEV_18133351923974
// MI455X (gfx1250) — matrix operand order repaired and checked by reading, not compiled, not run on hardware
//
#include <hip/hip_runtime.h>
#include <hip/hip_bf16.h>

typedef __attribute__((ext_vector_type(16))) _Float16 v16h;
typedef __attribute__((ext_vector_type(8)))  _Float16 v8h;
typedef __attribute__((ext_vector_type(8)))  float    v8f;
typedef __attribute__((ext_vector_type(4)))  float    v4f;
typedef __attribute__((ext_vector_type(2)))  float    v2f;

#define BEVH 128
#define BEVW 128
#define HW   (BEVH * BEVW)
#define C1   80            // hidden / embedding channels
#define KP   96            // K padded to 3 x 32 for wmma_f32_16x16x32_f16
#define COUT 128
#define NBATCH 4
#define DUMMY (NBATCH * HW)          // dummy bucket for invalid points

// ---------------------------------------------------------------------------
// Fragment builders (wave32, v_wmma_f32_16x16x32_f16; contiguous K order chosen by this code, not the ISA table order)
// A 16x32 f16: lanes 0-15 -> row M=lane, K in [0,16)
//              lanes 16-31 -> row M=lane-16, K in [16,32)
// B 32x16 f16: lane&15 -> column N, lanes>=16 hold K in [16,32); K packed 2/VGPR
// Both loaders read [row][KP] f16 LDS arrays (rows 192B apart, 16B aligned).
// ---------------------------------------------------------------------------
__device__ __forceinline__ v16h load_a_frag(const _Float16* __restrict__ row,
                                            int kk, int hs) {
    const v8h lo = *(const v8h*)(row + kk + hs * 16);
    const v8h hi = *(const v8h*)(row + kk + hs * 16 + 8);
    v16h a;
#pragma unroll
    for (int i = 0; i < 8; ++i) { a[i] = lo[i]; a[8 + i] = hi[i]; }
    return a;
}

__device__ __forceinline__ v16h load_b_frag(const _Float16* __restrict__ base,
                                            int n, int kk, int hs) {
    return *(const v16h*)(base + (size_t)n * KP + kk + hs * 16);
}

// ---------------------------------------------------------------------------
// Kernel C: zero the scatter accumulators (sums + dummy row + counts)
// ---------------------------------------------------------------------------
__global__ __launch_bounds__(256) void zero_ws_kernel(float* __restrict__ p, long n4) {
    long i = (long)blockIdx.x * 256 + threadIdx.x;
    if (i < n4) ((v4f*)p)[i] = (v4f){0.f, 0.f, 0.f, 0.f};
}

// ---------------------------------------------------------------------------
// Kernel A: point MLP (layer1 packed-f32 VALU, layer2 WMMA transposed) + scatter
//   Layer2 WMMA: A = W2^T (M = out channel), B = h (N = point).
//   C-frag: lane -> one point, vgpr r -> 8 consecutive channels
//   => one atomic base address per lane; nt*64+r*4 folds into immediates.
// Block: 256 threads (8 waves), 128 points. Grid: NBATCH * chunks.
// ---------------------------------------------------------------------------
__global__ __launch_bounds__(256)
void mlp_scatter_kernel(const float* __restrict__ points,
                        const float* __restrict__ W1, const float* __restrict__ b1,
                        const float* __restrict__ W2, const float* __restrict__ b2,
                        float* __restrict__ sums, float* __restrict__ cnt,
                        int Np, int chunks)
{
    __shared__ __align__(16) _Float16 hS[128 * KP];    // layer-1 activations, K-padded
    __shared__ __align__(16) _Float16 w2tS[C1 * KP];   // W2 transposed [n][k], K-padded
    __shared__ __align__(32) float b2S[C1];
    __shared__ __align__(8)  float w1S[4 * C1];
    __shared__ __align__(8)  float b1S[C1];
    __shared__ int   gS[128];                          // bucket id per point

    const int tid   = threadIdx.x;
    const int batch = blockIdx.x / chunks;
    const int chunk = blockIdx.x % chunks;
    const int p0    = chunk * 128;

    // ---- stage weights to LDS (one-time; keep code small) ---------------
#pragma unroll 2
    for (int i = tid; i < 4 * C1; i += 256) w1S[i] = W1[i];
    if (tid < C1) { b1S[tid] = b1[tid]; b2S[tid] = b2[tid]; }
#pragma unroll 2
    for (int i = tid; i < C1 * KP; i += 256) {
        const int n = i / KP, k = i % KP;
        w2tS[i] = (k < C1) ? (_Float16)W2[k * C1 + n] : (_Float16)0.f;
    }
    __syncthreads();

    // ---- layer 1: 2 threads per point, 48 channels each (incl. K pad) ---
    const int pl   = tid >> 1;
    const int half = tid & 1;
    const int p    = p0 + pl;
    const bool inrange = (p < Np);
    float f0 = 0.f, f1 = 0.f, f2 = 0.f, f3 = 0.f;
    if (inrange) {
        const v4f pv = *(const v4f*)(points + ((size_t)batch * Np + p) * 4);
        f0 = pv[0]; f1 = pv[1]; f2 = pv[2]; f3 = pv[3];
    }
    if (half == 0) {
        const int ix = (int)floorf((f0 + 50.f) * (BEVW / 100.f));
        const int iy = (int)floorf((f1 + 50.f) * (BEVH / 100.f));
        const bool valid = inrange && ix >= 0 && ix < BEVW && iy >= 0 && iy < BEVH;
        const int g = valid ? (batch * HW + iy * BEVW + ix) : DUMMY;
        gS[pl] = g;
        atomicAdd(&cnt[g], 1.0f);     // dummy bucket absorbs invalid, no branch
    }
    {
        const int cbase = half * 48;
        const v2f s0 = (v2f){f0, f0}, s1 = (v2f){f1, f1};
        const v2f s2 = (v2f){f2, f2}, s3 = (v2f){f3, f3};
        _Float16 hbuf[48];
#pragma unroll
        for (int j = 0; j < 24; ++j) {                // packed f32: v_pk_fma_f32
            const int c = cbase + j * 2;              // even, never straddles 80
            v2f a = (v2f){0.f, 0.f};
            if (c < C1 && inrange) {
                a  = *(const v2f*)(b1S + c);
                a += s0 * *(const v2f*)(w1S + 0 * C1 + c);
                a += s1 * *(const v2f*)(w1S + 1 * C1 + c);
                a += s2 * *(const v2f*)(w1S + 2 * C1 + c);
                a += s3 * *(const v2f*)(w1S + 3 * C1 + c);
                a[0] = fmaxf(a[0], 0.f);
                a[1] = fmaxf(a[1], 0.f);
            }
            hbuf[j * 2]     = (_Float16)a[0];
            hbuf[j * 2 + 1] = (_Float16)a[1];
        }
#pragma unroll
        for (int j = 0; j < 6; ++j)                   // 6 x ds_store_b128
            *(v8h*)(hS + pl * KP + cbase + j * 8) = *(const v8h*)(hbuf + j * 8);
    }
    __syncthreads();

    // ---- layer 2 (transposed): one wave per 16-point tile ---------------
    const int wv   = tid >> 5;
    const int lane = tid & 31;
    const int M    = lane & 15;       // A row -> channel-in-tile, B col -> point
    const int hs   = lane >> 4;
    const int prow = wv * 16;

    // B fragments: this lane's point row of h, all 3 K-steps (reused over nt)
    v16h bfr[3];
#pragma unroll
    for (int ks = 0; ks < 3; ++ks)
        bfr[ks] = load_b_frag(hS, prow + M, ks * 32, hs);

    // one atomic base per lane: bucket of this lane's point, channel base hs*8
    const unsigned eltoff = (unsigned)gS[prow + M] * C1 + hs * 8u;
    float* const abase = sums + eltoff;

#pragma unroll
    for (int nt = 0; nt < 5; ++nt) {
        v8f acc = {};
#pragma unroll
        for (int ks = 0; ks < 3; ++ks) {
            const v16h afr = load_a_frag(w2tS + (size_t)(nt * 16 + M) * KP,
                                         ks * 32, hs);
            acc = __builtin_amdgcn_wmma_f32_16x16x32_f16(
                false, afr, false, bfr[ks], (short)0, acc, false, false);
        }
        // channels nt*16 + hs*8 + r for r=0..7 : consecutive addresses
        const v8f b2v = *(const v8f*)(b2S + nt * 16 + hs * 8);
#pragma unroll
        for (int r = 0; r < 8; ++r)
            atomicAdd(abase + nt * 16 + r, fmaxf(acc[r] + b2v[r], 0.f));
    }
}

// ---------------------------------------------------------------------------
// Kernel B: mean -> 1x1 conv (WMMA) -> BN -> ReLU -> transposed NCHW store
// (non-transposed orientation: C-frag rows = consecutive pixels -> f32x4 stores)
// Block: 256 threads (8 waves), 128 BEV pixels. Grid: NBATCH*HW/128 = 512.
// ---------------------------------------------------------------------------
__global__ __launch_bounds__(256)
void bev_head_kernel(const float* __restrict__ Wp, const float* __restrict__ bp,
                     const float* __restrict__ gamma, const float* __restrict__ beta,
                     const float* __restrict__ rmean, const float* __restrict__ rvar,
                     const float* __restrict__ sums, const float* __restrict__ cnt,
                     float* __restrict__ out)
{
    __shared__ __align__(16) _Float16 wptS[COUT * KP];  // Wp transposed [n][k]
    __shared__ __align__(16) _Float16 mS[128 * KP];     // pixel means, K-padded
    __shared__ float scS[COUT];
    __shared__ float shS[COUT];

    const int tid = threadIdx.x;

#pragma unroll 2
    for (int i = tid; i < COUT * KP; i += 256) {
        const int n = i / KP, k = i % KP;
        wptS[i] = (k < C1) ? (_Float16)Wp[k * COUT + n] : (_Float16)0.f;
    }
    if (tid < COUT) {
        const float s = gamma[tid] * rsqrtf(rvar[tid] + 1e-5f);
        scS[tid] = s;
        shS[tid] = (bp[tid] - rmean[tid]) * s + beta[tid];
    }

    // means: 2 threads per pixel, 48 channels each, vectorized f32x4 loads
    const int pl   = tid >> 1;
    const int half = tid & 1;
    const size_t pg = (size_t)blockIdx.x * 128 + pl;    // global pixel in [0, B*HW)
    const float inv = 1.f / fmaxf(cnt[pg], 1.f);
    {
        const float* srow = sums + pg * C1;
        const int cbase = half * 48;
        _Float16 mbuf[48];
#pragma unroll
        for (int j = 0; j < 12; ++j) {                  // 4-wide chunks, no straddle
            const int c4 = cbase + j * 4;
            v4f v = (v4f){0.f, 0.f, 0.f, 0.f};
            if (c4 < C1) v = *(const v4f*)(srow + c4);
#pragma unroll
            for (int u = 0; u < 4; ++u) mbuf[j * 4 + u] = (_Float16)(v[u] * inv);
        }
#pragma unroll
        for (int j = 0; j < 6; ++j)                     // 6 x ds_store_b128
            *(v8h*)(mS + pl * KP + cbase + j * 8) = *(const v8h*)(mbuf + j * 8);
    }
    __syncthreads();

    const int wv   = tid >> 5;
    const int lane = tid & 31;
    const int M    = lane & 15;
    const int hs   = lane >> 4;
    const int prow = wv * 16;

    v16h afr[3];
#pragma unroll
    for (int ks = 0; ks < 3; ++ks)
        afr[ks] = load_a_frag(mS + (size_t)(prow + M) * KP, ks * 32, hs);

    const size_t pgbase = (size_t)blockIdx.x * 128 + prow;
    const int batch = (int)(pgbase >> 14);              // HW = 16384
    const int hw    = (int)(pgbase & (HW - 1));

#pragma unroll
    for (int nt = 0; nt < 8; ++nt) {
        v8f acc = {};
#pragma unroll
        for (int ks = 0; ks < 3; ++ks) {
            const v16h bfr = load_b_frag(wptS, nt * 16 + M, ks * 32, hs);
            acc = __builtin_amdgcn_wmma_f32_16x16x32_f16(
                false, afr[ks], false, bfr, (short)0, acc, false, false);
        }
        const int   ch = nt * 16 + M;
        const float s  = scS[ch];
        const float sh = shS[ch];
        float* orow = out + ((size_t)(batch * COUT + ch) << 14) + hw + hs * 8;
        v4f lo, hi;
#pragma unroll
        for (int r = 0; r < 4; ++r) {
            lo[r] = fmaxf(fmaf(acc[r],     s, sh), 0.f);
            hi[r] = fmaxf(fmaf(acc[r + 4], s, sh), 0.f);
        }
        *(v4f*)orow       = lo;
        *(v4f*)(orow + 4) = hi;
    }
}

// ---------------------------------------------------------------------------
extern "C" void kernel_launch(void* const* d_in, const int* in_sizes, int n_in,
                              void* d_out, int out_size, void* d_ws, size_t ws_size,
                              hipStream_t stream) {
    const float* points = (const float*)d_in[0];
    const float* W1     = (const float*)d_in[1];
    const float* b1     = (const float*)d_in[2];
    const float* W2     = (const float*)d_in[3];
    const float* b2     = (const float*)d_in[4];
    const float* Wp     = (const float*)d_in[5];
    const float* bp     = (const float*)d_in[6];
    const float* gamma  = (const float*)d_in[7];
    const float* beta   = (const float*)d_in[8];
    const float* rmean  = (const float*)d_in[9];
    const float* rvar   = (const float*)d_in[10];

    const int Np = in_sizes[0] / (NBATCH * 4);

    // workspace: sums[(B*HW+1) x 80] (+pad to 4), then cnt[B*HW+1] (+pad)
    const size_t sums_elems = ((size_t)(DUMMY + 1) * C1 + 3) & ~(size_t)3;
    const size_t cnt_elems  = ((size_t)(DUMMY + 1) + 3) & ~(size_t)3;
    float* sums = (float*)d_ws;
    float* cntp = sums + sums_elems;

    const long n4 = (long)(sums_elems + cnt_elems) / 4;
    zero_ws_kernel<<<(int)((n4 + 255) / 256), 256, 0, stream>>>(sums, n4);

    const int chunks = (Np + 127) / 128;
    mlp_scatter_kernel<<<NBATCH * chunks, 256, 0, stream>>>(
        points, W1, b1, W2, b2, sums, cntp, Np, chunks);

    bev_head_kernel<<<(NBATCH * HW) / 128, 256, 0, stream>>>(
        Wp, bp, gamma, beta, rmean, rvar, sums, cntp, (float*)d_out);
}
